// PCAMask_45432164057788
// MI455X (gfx1250) — matrix operand order repaired and checked by reading, not compiled, not run on hardware
//
#include <hip/hip_runtime.h>
#include <hip/hip_bf16.h>

// PCA rank-1 mask for MI455X (gfx1250), one workgroup (8 wave32 waves) per sample.
// Strategy: G = Xc^T Xc once per sample via v_wmma_f32_16x16x32_bf16 (L2-resident
// Xc), then 6 cheap 512x512 matvec power steps on bf16 G (L2-resident), then
// fp32 scores + minmax + quantile + outputs. Workspace: BT*C*C bf16 = 128 MiB.

#define BT 256
#define CCH 512
#define HW 576         // 24*24 tokens
#define NSTEPS 6       // NITER + 1
#define KEEPQ 0.6f
#define EPSN 1e-6f
#define LSTR 40        // staged LDS row stride in bf16 (80B: banks + 16B align)

typedef __bf16 v16bf __attribute__((ext_vector_type(16)));
typedef __bf16 v8bf  __attribute__((ext_vector_type(8)));
typedef __bf16 v2bf  __attribute__((ext_vector_type(2)));
typedef float  v8f   __attribute__((ext_vector_type(8)));

__global__ __launch_bounds__(256)
void pca_mask_kernel(const float* __restrict__ x,
                     const int* __restrict__ valid_mask,
                     float* __restrict__ out,
                     __bf16* __restrict__ G)
{
    const int b    = blockIdx.x;
    const int t    = threadIdx.x;
    const int lane = t & 31;
    const int wave = t >> 5;

    const float* __restrict__ xb = x + (size_t)b * CCH * HW;
    __bf16* __restrict__ Gb = G + (size_t)b * CCH * CCH;

    __shared__ float  mu[CCH];
    __shared__ float  vv[CCH];
    __shared__ float  sarr[HW];
    __shared__ __bf16 As[64 * LSTR];
    __shared__ __bf16 Bs[64 * LSTR];
    __shared__ float  red[16];
    __shared__ float  sc0, sc1;

    // ---------------- Phase 1: per-channel means (center across tokens) ----
    for (int c = wave; c < CCH; c += 8) {
        const float* row = xb + c * HW;
        float s = 0.f;
        for (int n = lane; n < HW; n += 32) s += row[n];
        #pragma unroll
        for (int o = 16; o; o >>= 1) s += __shfl_xor(s, o, 32);
        if (lane == 0) mu[c] = s * (1.0f / (float)HW);
    }
    __syncthreads();

    // ---------------- Phase 2: G = Xc^T Xc, bf16 WMMA, 64x64 super-tiles ---
    for (int I = 0; I < 8; ++I)
    for (int J = I; J < 8; ++J) {
        // 16 tiles of 16x16 per super-tile; each wave owns 2 tiles (same ti).
        const int tile0 = wave * 2;
        const int ti = tile0 >> 2;
        const int tj0 = tile0 & 3;
        const int tj1 = tj0 + 1;
        v8f acc0 = {};
        v8f acc1 = {};

        for (int ks = 0; ks < 18; ++ks) {      // K loop over N=576 in steps of 32
            const int n0 = ks * 32;
            __syncthreads();                   // protect LDS reuse
            // Cooperative stage: 64 rows x 32 K for A-block (rows I*64+) and
            // B-block (rows J*64+), centered, packed bf16 pairs -> ds_store_b32.
            // Coalesced: 16 consecutive lanes cover one 128B row segment.
            {
                const int kp = (t & 15) * 2;   // K pair
                const int rb = t >> 4;         // 0..15
                #pragma unroll
                for (int e = 0; e < 4; ++e) {
                    const int r  = e * 16 + rb;
                    const int ca = I * 64 + r;
                    const int cb = J * 64 + r;
                    const float2 fa = *(const float2*)&xb[ca * HW + n0 + kp];
                    const float2 fb = *(const float2*)&xb[cb * HW + n0 + kp];
                    const float ma = mu[ca];
                    const float mb2 = mu[cb];
                    v2bf pa, pb;
                    pa[0] = (__bf16)(fa.x - ma);  pa[1] = (__bf16)(fa.y - ma);
                    pb[0] = (__bf16)(fb.x - mb2); pb[1] = (__bf16)(fb.y - mb2);
                    *(v2bf*)&As[r * LSTR + kp] = pa;
                    *(v2bf*)&Bs[r * LSTR + kp] = pb;
                    if (ks + 1 < 18) {         // warm L2->WGP path for next slab
                        __builtin_prefetch(&xb[ca * HW + n0 + 32 + kp], 0, 0);
                    }
                }
            }
            __syncthreads();

            // A fragment (16x32 bf16, ISA 7.12.2: lane m rows, split K halves)
            v16bf af;
            {
                const int row = ti * 16 + (lane & 15);
                const int kb  = (lane < 16) ? 0 : 8;
                v8bf lo = *(const v8bf*)&As[row * LSTR + kb];        // K=kb..kb+7
                v8bf hi = *(const v8bf*)&As[row * LSTR + kb + 16];   // K=kb+16..kb+23
                #pragma unroll
                for (int e = 0; e < 8; ++e) { af[e] = lo[e]; af[8 + e] = hi[e]; }
            }
            // B fragments (32x16: lane holds one column = one Xc row, split K halves like A)
            v16bf bf0, bf1;
            {
                const int kb   = (lane < 16) ? 0 : 8;
                const int col0 = tj0 * 16 + (lane & 15);
                const int col1 = tj1 * 16 + (lane & 15);
                v8bf a0 = *(const v8bf*)&Bs[col0 * LSTR + kb];
                v8bf a1 = *(const v8bf*)&Bs[col0 * LSTR + kb + 16];
                v8bf c0 = *(const v8bf*)&Bs[col1 * LSTR + kb];
                v8bf c1 = *(const v8bf*)&Bs[col1 * LSTR + kb + 16];
                #pragma unroll
                for (int e = 0; e < 8; ++e) {
                    bf0[e] = a0[e]; bf0[8 + e] = a1[e];
                    bf1[e] = c0[e]; bf1[8 + e] = c1[e];
                }
            }
            acc0 = __builtin_amdgcn_wmma_f32_16x16x32_bf16(
                       false, af, false, bf0, (short)0, acc0, false, false);
            acc1 = __builtin_amdgcn_wmma_f32_16x16x32_bf16(
                       false, af, false, bf1, (short)0, acc1, false, false);
        }

        // Write out. C/D layout: VGPR r -> M = r (+8 for upper lanes), N = lane&15.
        // Mirror (transposed) block is contiguous along gi per lane -> one
        // global_store_b128 of 8 packed bf16. Direct block stays scalar b16.
        {
            const int n_  = lane & 15;
            const int mb  = (lane < 16) ? 0 : 8;
            const int gi0 = I * 64 + ti * 16 + mb;
            const int gj0 = J * 64 + tj0 * 16 + n_;
            const int gj1 = J * 64 + tj1 * 16 + n_;
            v8bf p0, p1;
            #pragma unroll
            for (int r = 0; r < 8; ++r) {
                p0[r] = (__bf16)acc0[r];
                p1[r] = (__bf16)acc1[r];
            }
            *(v8bf*)&Gb[gj0 * CCH + gi0] = p0;   // lower-triangle mirror, packed
            *(v8bf*)&Gb[gj1 * CCH + gi0] = p1;
            #pragma unroll
            for (int r = 0; r < 8; ++r) {        // upper-triangle direct, scattered
                Gb[(gi0 + r) * CCH + gj0] = p0[r];
                Gb[(gi0 + r) * CCH + gj1] = p1[r];
            }
        }
    }
    __threadfence();      // make this block's G stores visible to its own reads
    __syncthreads();

    // ---------------- Phase 3: power iteration v <- normalize(G v) ---------
    // Deterministic pseudo-random init (direction-only; SVD sign ambiguity).
    for (int c = t; c < CCH; c += 256) {
        unsigned u = (unsigned)c * 2654435761u + 0x9E3779B9u;
        u ^= u >> 13; u *= 2246822519u; u ^= u >> 16;
        vv[c] = ((float)(u & 0xFFFFFFu) * (1.0f / 8388608.0f)) - 1.0f;
    }
    __syncthreads();

    const int r0 = t, r1 = t + 256;
    const __bf16* __restrict__ g0 = Gb + (size_t)r0 * CCH;
    const __bf16* __restrict__ g1 = Gb + (size_t)r1 * CCH;
    for (int it = 0; it < NSTEPS; ++it) {
        float a0 = 0.f, a1 = 0.f;
        for (int c0 = 0; c0 < CCH; c0 += 8) {
            v8bf ga = *(const v8bf*)(g0 + c0);
            v8bf gb = *(const v8bf*)(g1 + c0);
            #pragma unroll
            for (int e = 0; e < 8; ++e) {
                const float vc = vv[c0 + e];
                a0 += (float)ga[e] * vc;
                a1 += (float)gb[e] * vc;
            }
        }
        float ss = a0 * a0 + a1 * a1;
        #pragma unroll
        for (int o = 16; o; o >>= 1) ss += __shfl_xor(ss, o, 32);
        if (lane == 0) red[wave] = ss;
        __syncthreads();
        if (t == 0) {
            float s = 0.f;
            for (int w = 0; w < 8; ++w) s += red[w];
            sc0 = 1.0f / (sqrtf(s) + 1e-12f);
        }
        __syncthreads();
        vv[r0] = a0 * sc0;
        vv[r1] = a1 * sc0;
        __syncthreads();
    }

    // ---------------- Phase 4: scores = Xc v, minmax, quantile, outputs ----
    const bool valid = (valid_mask[b] != 0);
    float s0 = 0.f, s1 = 0.f, s2 = 0.f;
    for (int c = 0; c < CCH; ++c) {
        const float vc = vv[c];
        const float m  = mu[c];
        const float* row = xb + c * HW;
        s0 += (row[t] - m) * vc;
        s1 += (row[t + 256] - m) * vc;
        if (t < 64) s2 += (row[t + 512] - m) * vc;
    }
    sarr[t] = s0; sarr[t + 256] = s1;
    if (t < 64) sarr[t + 512] = s2;

    // block min/max
    float mn = fminf(s0, s1), mx = fmaxf(s0, s1);
    if (t < 64) { mn = fminf(mn, s2); mx = fmaxf(mx, s2); }
    #pragma unroll
    for (int o = 16; o; o >>= 1) {
        mn = fminf(mn, __shfl_xor(mn, o, 32));
        mx = fmaxf(mx, __shfl_xor(mx, o, 32));
    }
    if (lane == 0) { red[wave] = mn; red[8 + wave] = mx; }
    __syncthreads();
    if (t == 0) {
        float gmn = red[0], gmx = red[8];
        for (int w = 1; w < 8; ++w) {
            gmn = fminf(gmn, red[w]);
            gmx = fmaxf(gmx, red[8 + w]);
        }
        sc0 = gmn;
        sc1 = 1.0f / (gmx - gmn + EPSN);
    }
    __syncthreads();
    for (int n = t; n < HW; n += 256) sarr[n] = (sarr[n] - sc0) * sc1;
    __syncthreads();

    // order-statistic quantile via exact ranking (ties broken by index)
    const float pos  = KEEPQ * (float)(HW - 1);
    const int   ilo  = (int)pos;
    const int   ihi  = (ilo + 1 < HW) ? ilo + 1 : HW - 1;
    const float frac = pos - (float)ilo;
    for (int n = t; n < HW; n += 256) {
        const float sn = sarr[n];
        int rank = 0;
        for (int j = 0; j < HW; ++j) {
            const float sj = sarr[j];
            rank += (sj < sn) || (sj == sn && j < n);
        }
        if (rank == ilo) sc0 = sn;
        if (rank == ihi) sc1 = sn;
    }
    __syncthreads();
    const float thr = sc0 + frac * (sc1 - sc0);

    float* __restrict__ mask_out  = out;
    float* __restrict__ score_out = out + (size_t)BT * HW;
    for (int n = t; n < HW; n += 256) {
        const float si = sarr[n];
        mask_out [(size_t)b * HW + n] = (valid && si >= thr) ? 1.0f : 0.0f;
        score_out[(size_t)b * HW + n] = valid ? si : 0.0f;
    }
}

extern "C" void kernel_launch(void* const* d_in, const int* in_sizes, int n_in,
                              void* d_out, int out_size, void* d_ws, size_t ws_size,
                              hipStream_t stream) {
    (void)in_sizes; (void)n_in; (void)out_size; (void)ws_size;
    const float* x     = (const float*)d_in[0];
    const int*   valid = (const int*)d_in[1];
    float*       out   = (float*)d_out;
    __bf16*      G     = (__bf16*)d_ws;   // BT*CCH*CCH bf16 = 128 MiB scratch
    pca_mask_kernel<<<dim3(BT), dim3(256), 0, stream>>>(x, valid, out, G);
}
